// AutoCorrelation_53996328845740
// MI455X (gfx1250) — compile-verified, matrix operand order checked by reading, not run on hardware
//
#include <hip/hip_runtime.h>
#include <stdint.h>

// ---------------- types ----------------
typedef __attribute__((ext_vector_type(16))) __bf16 v16bf;
typedef __attribute__((ext_vector_type(8)))  __bf16 bf8;
typedef __attribute__((ext_vector_type(8)))  float  v8f;
typedef __attribute__((ext_vector_type(4)))  unsigned int u32x4;
typedef __attribute__((ext_vector_type(8)))  int    i32x8;
typedef __attribute__((ext_vector_type(4)))  int    i32x4;

union Frag16 { v16bf v; bf8 h[2]; };

#define BATCH 2
#define NH 8
#define SEQ 2048
#define DH 64
#define EMB 512
#define NC 66     // real coefficients: 1 + 2*32
#define NCP 96    // padded K for 3x k32 wmma steps
#define NROWS (BATCH*NH*SEQ)

__device__ __forceinline__ __bf16 f2bf(float x) {
    union { float f; uint32_t u; } a; a.f = x;
    uint32_t r = a.u + 0x7FFFu + ((a.u >> 16) & 1u);
    uint16_t hbits = (uint16_t)(r >> 16);
    __bf16 o; __builtin_memcpy(&o, &hbits, 2); return o;
}
__device__ __forceinline__ float bf2f(__bf16 x) {
    uint16_t hbits; __builtin_memcpy(&hbits, &x, 2);
    union { uint32_t u; float f; } a; a.u = ((uint32_t)hbits) << 16; return a.f;
}

// TDM: load a [64 rows x 64B] bf16 tile (rows stride 4096B) into LDS.
// D# per CDNA5 ISA ch.8: group0 = count/lds/global/type, group1 = dims.
// This toolchain's builtin is the 6-arg form (g0, g1, g2, g3, g4, cpol).
__device__ __forceinline__ void tdm_load_vchunk(const __bf16* gptr, uint32_t lds_byte_addr) {
    uint64_t ga = (uint64_t)(size_t)gptr;
    u32x4 g0 = { 1u,                                   // count=1 (valid), user mode
                 lds_byte_addr,                        // lds_addr
                 (uint32_t)ga,                         // global_addr[31:0]
                 (uint32_t)((ga >> 32) & 0x01FFFFFFu)  // global_addr[56:32]
                   | 0x80000000u };                    // type=2 ("image")
    i32x8 g1 = { (int)(2u << 16),       // wg_mask=0, data_size=2 (4-byte units)
                 (int)(1024u << 16),    // tensor_dim0[15:0]=1024 dwords/row
                 (int)(64u << 16),      // tensor_dim0 hi=0 | tensor_dim1[15:0]=64
                 (int)(16u << 16),      // tensor_dim1 hi=0 | tile_dim0=16 dwords
                 64,                    // tile_dim1=64 rows, tile_dim2=0
                 1024,                  // tensor_dim0_stride lo32 = 1024 dwords
                 0, 0 };                // stride hi, dim1_stride = 0 (2D)
    i32x4 gz4 = { 0, 0, 0, 0 };
    i32x8 gz8 = { 0, 0, 0, 0, 0, 0, 0, 0 };
    __builtin_amdgcn_tensor_load_to_lds(g0, g1, gz4, gz4, gz8, 0);
}

// ---------------- K0: basis table BasisT[l][j], bf16, [2048][96] ----------------
__global__ __launch_bounds__(256) void k_basis(__bf16* __restrict__ BasisT) {
    int i = blockIdx.x * 256 + threadIdx.x;          // 2048*96 = 196608
    if (i >= SEQ * NCP) return;
    int l = i / NCP, j = i % NCP;
    float v;
    if (j == 0) v = 1.0f;
    else if (j < NC) {
        int f = (j + 1) >> 1;
        float ang = (float)f * (float)l * (3.14159265358979323846f / 1024.0f);
        v = (j & 1) ? cosf(ang) : sinf(ang);         // odd j: cos, even j: sin
    } else v = 0.0f;
    BasisT[i] = f2bf(v);
}

// ---------------- K1: DFT coeffs per row, head-mean folded out -------------
__global__ __launch_bounds__(288) void k_coeffs(const float* __restrict__ q,
                                                const float* __restrict__ k,
                                                __bf16* __restrict__ Cdiff) {
    __shared__ float qs[EMB], ks[EMB];
    __shared__ float cs[64], sn[64];
    __shared__ float cl[NH][NC];
    __shared__ float mn[NC];
    const int tid = threadIdx.x;
    const int bs  = blockIdx.x;                       // b*2048 + s
    const size_t base = (size_t)bs * EMB;
    for (int i = tid; i < EMB; i += 288) { qs[i] = q[base + i]; ks[i] = k[base + i]; }
    if (tid < 64) {
        float ang = (float)tid * (6.28318530717958647692f / 64.0f);
        cs[tid] = cosf(ang); sn[tid] = sinf(ang);
    }
    __syncthreads();
    if (tid < NH * 33) {
        int h = tid / 33, f = tid % 33;
        float Qr = 0.f, Qi = 0.f, Kr = 0.f, Ki = 0.f;
        for (int t = 0; t < 64; ++t) {
            int idx = (f * t) & 63;
            float c = cs[idx], s = sn[idx];
            float qq = qs[h * 64 + t], kk = ks[h * 64 + t];
            Qr += qq * c; Qi -= qq * s;
            Kr += kk * c; Ki -= kk * s;
        }
        float Xr = Qr * Kr + Qi * Ki;                 // Q * conj(K)
        float Xi = Qi * Kr - Qr * Ki;
        const float inv = 1.0f / 2048.0f;
        if (f == 0) cl[h][0] = Xr * inv;
        else { cl[h][2 * f - 1] = 2.0f * Xr * inv; cl[h][2 * f] = -2.0f * Xi * inv; }
    }
    __syncthreads();
    if (tid < NC) {
        float s = 0.f;
        for (int h = 0; h < NH; ++h) s += cl[h][tid];
        mn[tid] = s * 0.125f;
    }
    __syncthreads();
    const int b = bs >> 11, s = bs & (SEQ - 1);
    for (int i = tid; i < NH * NCP; i += 288) {
        int h = i / NCP, j = i % NCP;
        float v = (j < NC) ? (cl[h][j] - mn[j]) : 0.0f;
        Cdiff[((size_t)(b * NH + h) * SEQ + s) * NCP + j] = f2bf(v);
    }
}

// ---------------- K2: per-row softmax stats (max, 1/sumexp) ----------------
__global__ __launch_bounds__(256) void k_stats(const __bf16* __restrict__ Cdiff,
                                               const __bf16* __restrict__ BasisT,
                                               float* __restrict__ Mrow,
                                               float* __restrict__ invS) {
    __shared__ float csh[8][NC];
    const int tid = threadIdx.x, wv = tid >> 5, lane = tid & 31;
    const size_t row0 = (size_t)blockIdx.x * 8;
    for (int i = tid; i < 8 * NC; i += 256) {
        int w = i / NC, j = i % NC;
        csh[w][j] = bf2f(Cdiff[(row0 + w) * NCP + j]);
    }
    __syncthreads();
    float m = -3.0e38f, ssum = 0.0f;
    for (int l = lane; l < SEQ; l += 32) {
        const __bf16* bp = BasisT + (size_t)l * NCP;
        float acc = 0.0f;
        for (int j = 0; j < NC; ++j) acc += csh[wv][j] * bf2f(bp[j]);
        if (acc > m) { ssum = ssum * __expf(m - acc) + 1.0f; m = acc; }
        else         { ssum += __expf(acc - m); }
    }
    for (int off = 16; off >= 1; off >>= 1) {
        float m2 = __shfl_xor(m, off, 32);
        float s2 = __shfl_xor(ssum, off, 32);
        float M  = fmaxf(m, m2);
        ssum = ssum * __expf(m - M) + s2 * __expf(m2 - M);
        m = M;
    }
    if (lane == 0) { Mrow[row0 + wv] = m; invS[row0 + wv] = 1.0f / ssum; }
}

// ---------------- K2.5: pre-scale v into bf16 [bh][d][s] with 1/S folded ---
__global__ __launch_bounds__(256) void k_vprep(const float* __restrict__ values,
                                               const float* __restrict__ invS,
                                               __bf16* __restrict__ Vs) {
    int i = blockIdx.x * 256 + threadIdx.x;           // 16*64*2048 = 2,097,152
    if (i >= BATCH * NH * DH * SEQ) return;
    int s  = i & (SEQ - 1);
    int d  = (i >> 11) & (DH - 1);
    int bh = i >> 17;
    int b = bh >> 3, h = bh & 7;
    float v = values[((size_t)(b * SEQ + s)) * EMB + h * DH + d]
              * invS[(size_t)bh * SEQ + s];
    Vs[i] = f2bf(v);
}

// ---------------- K3: fused corr-WMMA -> exp -> output-WMMA ----------------
// grid: (SEQ/128, BATCH*NH), block 256 (8 waves); wave owns a 16-wide l tile.
// v chunks arrive via TDM (tensor_load_to_lds), double-buffered in LDS.
__global__ __launch_bounds__(256) void k_fused(const __bf16* __restrict__ Vs,
                                               const __bf16* __restrict__ Cdiff,
                                               const __bf16* __restrict__ BasisT,
                                               const float* __restrict__ Mrow,
                                               float* __restrict__ out) {
    __shared__ __align__(128) __bf16 vlds[2][DH * 32];   // 2 x 4 KB tiles
    const int bh   = blockIdx.y;                      // 0..15
    const int b    = bh >> 3, h = bh & 7;
    const int tid  = threadIdx.x;
    const int wv   = tid >> 5;
    const int lane = tid & 31;
    const int n     = lane & 15;                      // tile row/col index
    const int khalf = lane >> 4;                      // fragment K half
    const int moff  = (lane < 16) ? 0 : 8;            // C-layout row offset
    const int l0    = blockIdx.x * 128 + wv * 16;     // wave's l-tile base

    const __bf16* cd    = Cdiff + (size_t)bh * SEQ * NCP;
    const float*  Mr    = Mrow  + (size_t)bh * SEQ;
    const __bf16* vbase = Vs    + (size_t)bh * DH * SEQ;
    const uint32_t lds0 = (uint32_t)(size_t)&vlds[0][0];
    const uint32_t lds1 = (uint32_t)(size_t)&vlds[1][0];

    // Basis B-fragments: reused across the whole s loop (3 k-steps of 32)
    Frag16 bb[3];
    {
        const __bf16* bp = BasisT + (size_t)(l0 + n) * NCP;
        for (int ks = 0; ks < 3; ++ks) {
            bb[ks].h[0] = *(const bf8*)(bp + ks * 32 + khalf * 8);
            bb[ks].h[1] = *(const bf8*)(bp + ks * 32 + 16 + khalf * 8);
        }
    }

    const v8f vzero = {0.f, 0.f, 0.f, 0.f, 0.f, 0.f, 0.f, 0.f};
    v8f acc[4] = {vzero, vzero, vzero, vzero};

    if (wv == 0) tdm_load_vchunk(vbase, lds0);        // prologue: chunk 0

    for (int it = 0; it < SEQ / 32; ++it) {
        const int sb = it * 32;
        if (wv == 0) {
            if (it + 1 < SEQ / 32) {                  // issue chunk it+1, then
                tdm_load_vchunk(vbase + sb + 32, (it & 1) ? lds0 : lds1);
                __builtin_amdgcn_s_wait_tensorcnt(1); // chunk it done (in-order)
            } else {
                __builtin_amdgcn_s_wait_tensorcnt(0);
            }
        }
        __syncthreads();

        // prefetch next Cdiff tile rows into GL2
        if (sb + 32 < SEQ)
            __builtin_prefetch(cd + (size_t)(sb + 32 + n) * NCP, 0, 1);

        // corr tiles: two 16(s) x 16(l) WMMA accumulations over K=96
        v8f ct[2];
        for (int t2 = 0; t2 < 2; ++t2) {
            const __bf16* ap = cd + (size_t)(sb + t2 * 16 + n) * NCP;
            v8f c = vzero;
            for (int ks = 0; ks < 3; ++ks) {
                Frag16 a;
                a.h[0] = *(const bf8*)(ap + ks * 32 + khalf * 8);
                a.h[1] = *(const bf8*)(ap + ks * 32 + 16 + khalf * 8);
                c = __builtin_amdgcn_wmma_f32_16x16x32_bf16(
                        false, a.v, false, bb[ks].v, (short)0, c, false, false);
            }
            ct[t2] = c;
        }

        // exp(x - M[s]) packs straight into the 32x16 bf16 B-fragment:
        // C-layout (lane = col, 8-row chunks) == B-fragment k-chunk layout.
        Frag16 w;
        for (int r = 0; r < 8; ++r) {
            float m0 = Mr[sb + moff + r];
            float m1 = Mr[sb + 16 + moff + r];
            w.v[r]     = f2bf(__expf(ct[0][r] - m0));
            w.v[8 + r] = f2bf(__expf(ct[1][r] - m1));
        }

        // output GEMM: 4 d-tiles of 16, A = v/S tile from LDS (TDM-staged)
        const __bf16* vb = &vlds[it & 1][0];
        for (int dt = 0; dt < 4; ++dt) {
            Frag16 a;
            const __bf16* vp = vb + (dt * 16 + n) * 32;
            a.h[0] = *(const bf8*)(vp + khalf * 8);
            a.h[1] = *(const bf8*)(vp + 16 + khalf * 8);
            acc[dt] = __builtin_amdgcn_wmma_f32_16x16x32_bf16(
                          false, a.v, false, w.v, (short)0, acc[dt], false, false);
        }
        __syncthreads();   // all reads of buf[it&1] done before its reuse
    }

    // out flat per reference reshape: [B][d][H][L] -> (B, L, H*d)
    const size_t ob = (size_t)b * (SEQ * EMB) + (size_t)h * SEQ + (l0 + n);
    for (int dt = 0; dt < 4; ++dt)
        for (int r = 0; r < 8; ++r) {
            int dd = dt * 16 + moff + r;
            out[ob + (size_t)dd * (NH * SEQ)] = acc[dt][r];
        }
}

// ---------------- launcher ----------------
extern "C" void kernel_launch(void* const* d_in, const int* in_sizes, int n_in,
                              void* d_out, int out_size, void* d_ws, size_t ws_size,
                              hipStream_t stream) {
    const float* q = (const float*)d_in[0];
    const float* k = (const float*)d_in[1];
    const float* v = (const float*)d_in[2];
    float* out = (float*)d_out;

    uint8_t* ws = (uint8_t*)d_ws;
    const size_t szBasis = (size_t)SEQ * NCP * 2;              // 393,216
    const size_t szCdiff = (size_t)NROWS * NCP * 2;            // 6,291,456
    const size_t szM     = (size_t)NROWS * 4;                  // 131,072
    __bf16* BasisT = (__bf16*)ws;
    __bf16* Cdiff  = (__bf16*)(ws + szBasis);
    float*  Mrow   = (float*)(ws + szBasis + szCdiff);
    float*  invS   = (float*)(ws + szBasis + szCdiff + szM);
    __bf16* Vs     = (__bf16*)(ws + szBasis + szCdiff + 2 * szM);  // 4 MB

    k_basis <<<(SEQ * NCP + 255) / 256, 256, 0, stream>>>(BasisT);
    k_coeffs<<<BATCH * SEQ, 288, 0, stream>>>(q, k, Cdiff);
    k_stats <<<NROWS / 8, 256, 0, stream>>>(Cdiff, BasisT, Mrow, invS);
    k_vprep <<<(BATCH * NH * DH * SEQ) / 256, 256, 0, stream>>>(v, invS, Vs);
    dim3 grid(SEQ / 128, BATCH * NH);
    k_fused <<<grid, 256, 0, stream>>>(Vs, Cdiff, BasisT, Mrow, out);
}
